// SlotQualityEstimator_76948634075335
// MI455X (gfx1250) — compile-verified, matrix operand order checked by reading, not run on hardware
//
#include <hip/hip_runtime.h>
#include <hip/hip_bf16.h>
#include <math.h>

typedef __attribute__((ext_vector_type(16))) _Float16 v16h;
typedef __attribute__((ext_vector_type(8)))  float    v8f;
typedef __attribute__((ext_vector_type(4)))  float    v4f;

#define B_DIM 64
#define K_DIM 16
#define N_DIM 65536
#define D_DIM 256
#define H_DIM 64
#define CHUNKS 16            // N split into 16 chunks of 4096
#define RED_THREADS 256
#define VEC 4                // floats per lane per load (b128)

// ---------------------------------------------------------------------------
// Kernel 0: zero the workspace accumulators (win[B*K], tot[B*K])
// ---------------------------------------------------------------------------
__global__ void zero_ws_kernel(float* __restrict__ ws, int n) {
    int i = blockIdx.x * blockDim.x + threadIdx.x;
    if (i < n) ws[i] = 0.0f;
}

// ---------------------------------------------------------------------------
// Kernel 1: streaming reduction over attention_maps (memory-bound, 256 MB).
// Each block handles one (b, n-chunk). Per thread-iteration: 16 non-temporal
// global_load_b128 (4 consecutive n-columns x 16 k) -> component-wise max
// across k -> accumulate per-k totals (horizontal add) and winner mass.
// Wave butterfly reduce -> LDS atomics -> one global atomicAdd per (b,k).
// ---------------------------------------------------------------------------
__global__ void __launch_bounds__(RED_THREADS)
attn_reduce_kernel(const float* __restrict__ attn,
                   float* __restrict__ ws_win,
                   float* __restrict__ ws_tot) {
    const int b     = blockIdx.y;
    const int chunk = blockIdx.x;
    const int tid   = threadIdx.x;

    const float* base = attn + (size_t)b * K_DIM * N_DIM;

    float tot[K_DIM];
    float win[K_DIM];
#pragma unroll
    for (int k = 0; k < K_DIM; ++k) { tot[k] = 0.0f; win[k] = 0.0f; }

    // chunk covers 4096 n; each pass covers 256 threads * 4 n = 1024
    const int n0 = chunk * (N_DIM / CHUNKS) + tid * VEC;
#pragma unroll 1
    for (int it = 0; it < (N_DIM / CHUNKS) / (RED_THREADS * VEC); ++it) {
        const int n = n0 + it * RED_THREADS * VEC;
        v4f a[K_DIM];
#pragma unroll
        for (int k = 0; k < K_DIM; ++k) {
            a[k] = __builtin_nontemporal_load(
                       (const v4f*)(base + (size_t)k * N_DIM + n));
        }
        // component-wise max over k (4 independent n-columns)
        v4f m = a[0];
#pragma unroll
        for (int k = 1; k < K_DIM; ++k) {
#pragma unroll
            for (int c = 0; c < VEC; ++c) m[c] = fmaxf(m[c], a[k][c]);
        }
#pragma unroll
        for (int k = 0; k < K_DIM; ++k) {
            tot[k] += ((a[k][0] + a[k][1]) + (a[k][2] + a[k][3]));
#pragma unroll
            for (int c = 0; c < VEC; ++c) {
                win[k] += (a[k][c] >= m[c]) ? a[k][c] : 0.0f;
            }
        }
    }

    // wave32 butterfly reduction of all 32 accumulators
#pragma unroll
    for (int k = 0; k < K_DIM; ++k) {
#pragma unroll
        for (int off = 16; off >= 1; off >>= 1) {
            tot[k] += __shfl_xor(tot[k], off, 32);
            win[k] += __shfl_xor(win[k], off, 32);
        }
    }

    __shared__ float s_win[K_DIM];
    __shared__ float s_tot[K_DIM];
    if (tid < K_DIM) { s_win[tid] = 0.0f; s_tot[tid] = 0.0f; }
    __syncthreads();

    if ((tid & 31) == 0) {           // one lane per wave
#pragma unroll
        for (int k = 0; k < K_DIM; ++k) {
            atomicAdd(&s_tot[k], tot[k]);
            atomicAdd(&s_win[k], win[k]);
        }
    }
    __syncthreads();

    if (tid < K_DIM) {
        atomicAdd(&ws_win[b * K_DIM + tid], s_win[tid]);
    } else if (tid < 2 * K_DIM) {
        atomicAdd(&ws_tot[b * K_DIM + (tid - K_DIM)], s_tot[tid - K_DIM]);
    }
}

// ---------------------------------------------------------------------------
// Kernel 2: tiny MLP via WMMA + final combine. One wave (32 threads) per b.
// H = relu(slots[16x256] @ W1[256x64] + b1) via 8 K-steps x 4 N-tiles of
// v_wmma_f32_16x16x32_f16. Then second layer (dot with W2) via a 16-lane
// shuffle reduction, sigmoid, blend with attn_quality and confidence.
// ---------------------------------------------------------------------------
__global__ void __launch_bounds__(32)
mlp_combine_kernel(const float* __restrict__ slots,
                   const float* __restrict__ conf,
                   const float* __restrict__ W1,
                   const float* __restrict__ b1,
                   const float* __restrict__ W2,
                   const float* __restrict__ b2,
                   const float* __restrict__ ws_win,
                   const float* __restrict__ ws_tot,
                   float* __restrict__ out) {
    const int b    = blockIdx.x;
    const int lane = threadIdx.x;          // 0..31
    const int half = lane >> 4;            // 0 or 1 (K-group / row-half)
    const int l16  = lane & 15;            // row of A / col of B,C,D

    v8f acc0 = {}, acc1 = {}, acc2 = {}, acc3 = {};

#pragma unroll
    for (int kk = 0; kk < D_DIM / 32; ++kk) {
        // ---- A tile: slots rows 16*b..16*b+15, K slice kk*32..kk*32+31 ----
        // ISA 16-bit A 16x32 layout: lane l16 holds row M=l16; elements 0..7
        // are K = base..base+7, elements 8..15 are K = base+16..base+23,
        // where base = kk*32 + half*8.
        const float* srow = slots + ((size_t)(b * K_DIM + l16)) * D_DIM
                                  + kk * 32 + half * 8;
        v16h A;
#pragma unroll
        for (int i = 0; i < 16; ++i) {
            A[i] = (_Float16)srow[(i < 8) ? i : (i + 8)];
        }

        // ---- B tiles: W1[256x64] column-major striping per lane ----
        // lane holds column N = n*16 + l16; element i -> K row
        // kk*32 + half*8 + ((i<8)? i : i+8)
        const float* wbase = W1 + (size_t)(kk * 32 + half * 8) * H_DIM + l16;
#pragma unroll
        for (int n = 0; n < 4; ++n) {
            v16h Bv;
            const float* wcol = wbase + n * 16;
#pragma unroll
            for (int i = 0; i < 16; ++i) {
                const int krow = (i < 8) ? i : (i + 8);
                Bv[i] = (_Float16)wcol[(size_t)krow * H_DIM];
            }
            v8f c = (n == 0) ? acc0 : (n == 1) ? acc1 : (n == 2) ? acc2 : acc3;
            c = __builtin_amdgcn_wmma_f32_16x16x32_f16(
                    false, A, false, Bv, (short)0, c, false, false);
            if      (n == 0) acc0 = c;
            else if (n == 1) acc1 = c;
            else if (n == 2) acc2 = c;
            else             acc3 = c;
        }
    }

    // ---- bias + relu + second layer partial: lane holds cols j=n*16+l16,
    //      rows M = r + 8*half (r = 0..7) per C/D layout ----
    float part[8];
#pragma unroll
    for (int r = 0; r < 8; ++r) part[r] = 0.0f;

#pragma unroll
    for (int n = 0; n < 4; ++n) {
        const int j = n * 16 + l16;
        const float bv  = b1[j];
        const float w2v = W2[j];
        const v8f c = (n == 0) ? acc0 : (n == 1) ? acc1 : (n == 2) ? acc2 : acc3;
#pragma unroll
        for (int r = 0; r < 8; ++r) {
            float h = c[r] + bv;
            h = fmaxf(h, 0.0f);
            part[r] += h * w2v;
        }
    }

    // reduce across the 16 lanes of each half-wave (cols 0..63 for this half)
#pragma unroll
    for (int r = 0; r < 8; ++r) {
#pragma unroll
        for (int off = 8; off >= 1; off >>= 1) {
            part[r] += __shfl_xor(part[r], off, 32);
        }
    }

    if (l16 == 0) {
        const float b2v = b2[0];
#pragma unroll
        for (int r = 0; r < 8; ++r) {
            const int row = b * K_DIM + r + 8 * half;   // (b, k) flat index
            const float learned = 1.0f / (1.0f + __expf(-(part[r] + b2v)));
            const float w = ws_win[row];
            const float t = ws_tot[row];
            const float attnq = w / (t + 1e-8f);
            float q = 0.4f * attnq + 0.4f * learned;
            out[row] = q * 0.2f + conf[row] * 0.8f;
        }
    }
}

// ---------------------------------------------------------------------------
extern "C" void kernel_launch(void* const* d_in, const int* in_sizes, int n_in,
                              void* d_out, int out_size, void* d_ws, size_t ws_size,
                              hipStream_t stream) {
    const float* slots = (const float*)d_in[0];   // [B,K,D]
    const float* attn  = (const float*)d_in[1];   // [B,K,N]
    const float* conf  = (const float*)d_in[2];   // [B,K]
    const float* W1    = (const float*)d_in[3];   // [D,H]
    const float* b1    = (const float*)d_in[4];   // [H]
    const float* W2    = (const float*)d_in[5];   // [H,1]
    const float* b2    = (const float*)d_in[6];   // [1]
    float*       out   = (float*)d_out;           // [B,K]

    float* ws_win = (float*)d_ws;                 // [B*K]
    float* ws_tot = ws_win + B_DIM * K_DIM;       // [B*K]

    const int nacc = 2 * B_DIM * K_DIM;
    zero_ws_kernel<<<(nacc + 255) / 256, 256, 0, stream>>>(ws_win, nacc);

    dim3 grid1(CHUNKS, B_DIM);
    attn_reduce_kernel<<<grid1, RED_THREADS, 0, stream>>>(attn, ws_win, ws_tot);

    mlp_combine_kernel<<<B_DIM, 32, 0, stream>>>(slots, conf, W1, b1, W2, b2,
                                                 ws_win, ws_tot, out);
}
